// AnswerSynthesisEncoder_12987981103235
// MI455X (gfx1250) — compile-verified, matrix operand order checked by reading, not run on hardware
//
#include <hip/hip_runtime.h>
#include <hip/hip_bf16.h>
#include <math.h>

// ---------------------------------------------------------------------------
// Problem constants
//   H = 1024, Hh = 1025, 3*Hh = 3075
//   tokens: 512 query + 4096 passage = 4608 (query first, matching gh rows)
// Padded dims: Kp = 1056 (33*32, WMMA K-steps of 32), Np = 3136 per dir (49*64)
// ---------------------------------------------------------------------------
#define HDIM   1024
#define HH     1025
#define NG     3075          // 3*Hh
#define TQ     512
#define TP     4096
#define TOKS   4608
#define KP     1056
#define NP     3136
#define JSTRIDE 1056         // padded j stride for transposed Wih
#define NBLK_SCAN 65         // ceil(2050/32) outputs, 32 outputs per block

typedef __attribute__((ext_vector_type(16))) __bf16 v16bf;
typedef __attribute__((ext_vector_type(8)))  float  v8f;

union FragAB { v16bf v; unsigned u[8]; };

__device__ __forceinline__ unsigned short f2bf(float f) {
  unsigned u = __float_as_uint(f);
  unsigned r = u + 0x7FFFu + ((u >> 16) & 1u);   // round-to-nearest-even
  return (unsigned short)(r >> 16);
}

// ---------------------------------------------------------------------------
// Workspace layout (bytes, all 256B aligned)
// ---------------------------------------------------------------------------
#define BAR_OFF  ((size_t)0)                               // 128 u32 barrier state
#define X_OFF    ((size_t)512)                             // 2 seq * 2 buf * 2 (a,b) * 1024 f32
#define A_OFF    (X_OFF + (size_t)2*2*2*1024*4)            // bf16 h0: [tok][dir][KP]
#define B_OFF    (A_OFF + (size_t)TOKS*2*KP*2)             // bf16 Whh: [dir][NP][KP]
#define WT_OFF   (B_OFF + (size_t)2*NP*KP*2)               // f32 Wih^T: [dir][3][1024][JSTRIDE]
#define GH_OFF   (WT_OFF + (size_t)2*3*1024*JSTRIDE*4)     // f32 gh: [tok][dir][NP]

// ---------------------------------------------------------------------------
// Init: zero x ping-pong buffers + barrier state (ws is poisoned to 0xAA)
// ---------------------------------------------------------------------------
__global__ void k_init(float* xbuf, unsigned* bar) {
  int idx = blockIdx.x * 256 + threadIdx.x;
  if (idx < 8192) xbuf[idx] = 0.f;
  if (idx < 128)  bar[idx] = 0u;
}

// ---------------------------------------------------------------------------
// Build bf16 A = h0[tok][dir][k]; h0 row d0 = e[0:1025], d1 = e[1025:2048],f_s,f_e
// ---------------------------------------------------------------------------
__global__ void k_build_A(const float* __restrict__ e_q, const float* __restrict__ e_p,
                          const float* __restrict__ f_s, const float* __restrict__ f_e,
                          unsigned short* __restrict__ Abf) {
  size_t idx = (size_t)blockIdx.x * 256 + threadIdx.x;  // TOKS*2*KP total
  int k = (int)(idx % KP);
  size_t md = idx / KP;
  int d = (int)(md & 1);
  size_t m = md >> 1;
  const float* e; float fs, fe;
  if (m < TQ) { e = e_q + m * 2048; fs = 0.f; fe = 0.f; }
  else        { size_t p = m - TQ; e = e_p + p * 2048; fs = f_s[p]; fe = f_e[p]; }
  float v = 0.f;
  if (d == 0) { if (k < HH) v = e[k]; }
  else        { if (k < HH - 2) v = e[HH + k];
                else if (k == HH - 2) v = fs;
                else if (k == HH - 1) v = fe; }
  Abf[idx] = f2bf(v);
}

// B = bf16 copy of Whh (row n = output, col k), zero-padded to [NP][KP]
__global__ void k_build_B(const float* __restrict__ Whh_f, const float* __restrict__ Whh_b,
                          unsigned short* __restrict__ Bbf) {
  size_t idx = (size_t)blockIdx.x * 256 + threadIdx.x;  // 2*NP*KP total
  int k = (int)(idx % KP);
  size_t nd = idx / KP;
  int n = (int)(nd % NP);
  int d = (int)(nd / NP);
  const float* W = d ? Whh_b : Whh_f;
  float v = (n < NG && k < HH) ? W[(size_t)n * HH + k] : 0.f;
  Bbf[idx] = f2bf(v);
}

// Transposed Wih for coalesced GEMV: WT[((d*3+g)*1024 + k)*JSTRIDE + j]
__global__ void k_build_WT(const float* __restrict__ Wih_f, const float* __restrict__ Wih_b,
                           float* __restrict__ WT) {
  size_t idx = (size_t)blockIdx.x * 256 + threadIdx.x;  // 2*3*1024*JSTRIDE total
  int j = (int)(idx % JSTRIDE);
  size_t rest = idx / JSTRIDE;
  int k = (int)(rest % HDIM);
  size_t gd = rest / HDIM;
  int g = (int)(gd % 3);
  int d = (int)(gd / 3);
  const float* W = d ? Wih_b : Wih_f;
  WT[idx] = (j < HH) ? W[((size_t)g * HH + j) * HDIM + k] : 0.f;
}

// ---------------------------------------------------------------------------
// Phase 1 GEMM: gh[tok][d][n] = sum_k h0[tok][d][k] * Whh_d[n][k] + bhh_d[n]
// Block tile 128x64, 8 waves (4x2), each wave 32x32 = 2x2 WMMA 16x16 tiles.
// A frag (16x32 bf16): lane l<16 holds M=l, K={8kh..8kh+7, 16+8kh..23+8kh},
// kh = l>>4 — two contiguous 16B chunks per lane. B frag mirrors with N=l&15.
// ---------------------------------------------------------------------------
__global__ void __launch_bounds__(256)
k_gemm_gh(const unsigned short* __restrict__ Abf, const unsigned short* __restrict__ Bbf,
          const float* __restrict__ bhh_f, const float* __restrict__ bhh_b,
          float* __restrict__ gh) {
  __shared__ __align__(16) unsigned short As[128 * 32];
  __shared__ __align__(16) unsigned short Bs[64 * 32];

  int bn = blockIdx.x;            // 0..97 (49 per dir)
  int bm = blockIdx.y;            // 0..35
  int d  = bn >= 49;
  int n0 = (bn - d * 49) * 64;
  int m0 = bm * 128;
  int tid = threadIdx.x, lane = tid & 31, wave = tid >> 5;
  int wm = wave >> 1, wn = wave & 1;
  int kh = lane >> 4, ml = lane & 15;

  v8f zero = {0.f,0.f,0.f,0.f,0.f,0.f,0.f,0.f};
  v8f acc[2][2];
  acc[0][0] = zero; acc[0][1] = zero; acc[1][0] = zero; acc[1][1] = zero;

  for (int kc = 0; kc < KP / 32; ++kc) {
    // stage A tile: 128x32 halves (2 uint4 per thread)
    #pragma unroll
    for (int i = 0; i < 2; ++i) {
      int idx = tid + i * 256;
      int row = idx >> 2, c8 = idx & 3;
      *(uint4*)&As[row * 32 + c8 * 8] =
        *(const uint4*)&Abf[((size_t)(m0 + row) * 2 + d) * KP + kc * 32 + c8 * 8];
    }
    // stage B tile: 64x32 halves (1 uint4 per thread), n-major
    {
      int row = tid >> 2, c8 = tid & 3;
      *(uint4*)&Bs[row * 32 + c8 * 8] =
        *(const uint4*)&Bbf[((size_t)d * NP + n0 + row) * KP + kc * 32 + c8 * 8];
    }
    __syncthreads();

    FragAB a[2], b[2];
    #pragma unroll
    for (int ti = 0; ti < 2; ++ti) {
      int r = wm * 32 + ti * 16 + ml;
      *(uint4*)&a[ti].u[0] = *(const uint4*)&As[r * 32 + kh * 8];
      *(uint4*)&a[ti].u[4] = *(const uint4*)&As[r * 32 + 16 + kh * 8];
    }
    #pragma unroll
    for (int tj = 0; tj < 2; ++tj) {
      int c = wn * 32 + tj * 16 + ml;
      *(uint4*)&b[tj].u[0] = *(const uint4*)&Bs[c * 32 + kh * 8];
      *(uint4*)&b[tj].u[4] = *(const uint4*)&Bs[c * 32 + 16 + kh * 8];
    }
    #pragma unroll
    for (int ti = 0; ti < 2; ++ti)
      #pragma unroll
      for (int tj = 0; tj < 2; ++tj)
        acc[ti][tj] = __builtin_amdgcn_wmma_f32_16x16x32_bf16(
            false, a[ti].v, false, b[tj].v, (short)0, acc[ti][tj], false, false);
    __syncthreads();
  }

  const float* bhh = d ? bhh_b : bhh_f;
  // C layout: VGPR r, lane l -> M = r + 8*(l>>4), N = l&15
  #pragma unroll
  for (int ti = 0; ti < 2; ++ti)
    #pragma unroll
    for (int tj = 0; tj < 2; ++tj)
      #pragma unroll
      for (int r = 0; r < 8; ++r) {
        int m  = m0 + wm * 32 + ti * 16 + r + (kh << 3);
        int nc = n0 + wn * 32 + tj * 16 + ml;
        float bias = (nc < NG) ? bhh[nc] : 0.f;
        gh[((size_t)m * 2 + d) * NP + nc] = acc[ti][tj][r] + bias;
      }
}

// ---------------------------------------------------------------------------
// Device-scope two-phase barrier (sense via generation counter).
// bar[0] = arrive count, bar[64] = generation (separate cache lines).
// ---------------------------------------------------------------------------
__device__ __forceinline__ void gridBarrier(unsigned* bar, unsigned nblk) {
  __threadfence();          // each wave drains its own global stores to device scope
  __syncthreads();
  if (threadIdx.x == 0) {
    unsigned g = __atomic_load_n(bar + 64, __ATOMIC_RELAXED);
    unsigned prev = __atomic_fetch_add(bar, 1u, __ATOMIC_ACQ_REL);
    if (prev == nblk - 1u) {
      __atomic_store_n(bar, 0u, __ATOMIC_RELAXED);
      __atomic_store_n(bar + 64, g + 1u, __ATOMIC_RELEASE);
    } else {
      while (__atomic_load_n(bar + 64, __ATOMIC_ACQUIRE) == g)
        __builtin_amdgcn_s_sleep(2);
    }
  }
  __syncthreads();
}

// ---------------------------------------------------------------------------
// Phase 2: persistent scan kernel. 65 blocks * 256 threads.
// Each block owns 32 consecutive outputs o = (d,j) of the 2050 per step;
// 8-way K-split per output (s = tid>>5, j-lane = tid&31 -> coalesced WT reads).
// x is staged to LDS once per step (broadcast LDS reads in the GEMV loop).
// y -> x hand-off: d0/j<1024 -> xa[j]; d0/j==1024 -> xb[0]; d1/j<1023 -> xb[j+1];
// next step reads x[i] = xa[i] + xb[i].
// ---------------------------------------------------------------------------
__global__ void __launch_bounds__(256)
k_scan(const float* __restrict__ WT, const float* __restrict__ gh,
       const float* __restrict__ bih_f, const float* __restrict__ bih_b,
       const float* __restrict__ e_q, const float* __restrict__ e_p,
       const float* __restrict__ f_s, const float* __restrict__ f_e,
       float* __restrict__ out, float* __restrict__ xbuf, unsigned* bar) {
  __shared__ float xs[HDIM];
  __shared__ float red[8][32][3];

  int tid = threadIdx.x;
  int jl = tid & 31, s = tid >> 5;
  int o = blockIdx.x * 32 + jl;
  bool valid = (o < 2 * HH);
  int d = 0, j = 0;
  if (valid) { d = o / HH; j = o - d * HH; }
  const float* bih = d ? bih_b : bih_f;

  // per-thread weight stream bases (constant across all steps)
  size_t wb = ((size_t)(d * 3 + 0) * HDIM + (size_t)s * 128) * JSTRIDE + j;
  const float* wr0 = WT + wb;
  const float* wz0 = wr0 + (size_t)HDIM * JSTRIDE;
  const float* wn0 = wz0 + (size_t)HDIM * JSTRIDE;

  for (int seq = 0; seq < 2; ++seq) {
    int T = seq ? TP : TQ;
    int gbase = seq ? TQ : 0;
    size_t outbase = seq ? (size_t)TQ * 2048 : 0;
    float* xA[2]; float* xB[2];
    xA[0] = xbuf + ((size_t)(seq * 2 + 0) * 2 + 0) * HDIM;
    xB[0] = xbuf + ((size_t)(seq * 2 + 0) * 2 + 1) * HDIM;
    xA[1] = xbuf + ((size_t)(seq * 2 + 1) * 2 + 0) * HDIM;
    xB[1] = xbuf + ((size_t)(seq * 2 + 1) * 2 + 1) * HDIM;

    for (int t = 0; t < T; ++t) {
      int cur = t & 1, nxt = cur ^ 1;
      // stage x = xa + xb into LDS
      for (int i = tid; i < HDIM; i += 256) xs[i] = xA[cur][i] + xB[cur][i];
      if (tid < 32 && valid)  // warm gh row for this step while waiting
        __builtin_prefetch(gh + ((size_t)(gbase + t) * 2 + d) * NP + j, 0, 1);
      __syncthreads();

      float ar = 0.f, az = 0.f, an = 0.f;
      if (valid) {
        const float* pr = wr0;
        const float* pz = wz0;
        const float* pn = wn0;
        int kb = s * 128;
        #pragma unroll 4
        for (int kk = 0; kk < 128; ++kk) {
          float xv = xs[kb + kk];
          ar = fmaf(pr[0], xv, ar);
          az = fmaf(pz[0], xv, az);
          an = fmaf(pn[0], xv, an);
          pr += JSTRIDE; pz += JSTRIDE; pn += JSTRIDE;
        }
      }
      red[s][jl][0] = ar; red[s][jl][1] = az; red[s][jl][2] = an;
      __syncthreads();

      if (tid < 32 && valid) {
        float sr = 0.f, sz = 0.f, sn = 0.f;
        #pragma unroll
        for (int ss = 0; ss < 8; ++ss) {
          sr += red[ss][jl][0]; sz += red[ss][jl][1]; sn += red[ss][jl][2];
        }
        const float* ghrow = gh + ((size_t)(gbase + t) * 2 + d) * NP;
        float gir = sr + bih[j];
        float giz = sz + bih[HH + j];
        float gin = sn + bih[2 * HH + j];
        float hr = ghrow[j], hz = ghrow[HH + j], hn = ghrow[2 * HH + j];
        float r = 1.f / (1.f + expf(-(gir + hr)));
        float z = 1.f / (1.f + expf(-(giz + hz)));
        float n = tanhf(gin + r * hn);
        // h0 value (fp32, straight from inputs)
        float h0v;
        if (seq == 0) {
          const float* e = e_q + (size_t)t * 2048;
          h0v = (d == 0) ? e[j] : ((j < HH - 2) ? e[HH + j] : 0.f);
        } else {
          const float* e = e_p + (size_t)t * 2048;
          h0v = (d == 0) ? e[j]
                         : ((j < HH - 2) ? e[HH + j]
                                         : (j == HH - 2 ? f_s[t] : f_e[t]));
        }
        float y = (1.f - z) * n + z * h0v;
        if (j < HDIM) out[outbase + (size_t)t * 2048 + (size_t)d * HDIM + j] = y;
        if (d == 0) { if (j < HDIM) xA[nxt][j] = y; else xB[nxt][0] = y; }
        else if (j < HDIM - 1) xB[nxt][j + 1] = y;
      }
      gridBarrier(bar, NBLK_SCAN);
    }
  }
}

// ---------------------------------------------------------------------------
extern "C" void kernel_launch(void* const* d_in, const int* in_sizes, int n_in,
                              void* d_out, int out_size, void* d_ws, size_t ws_size,
                              hipStream_t stream) {
  (void)in_sizes; (void)n_in; (void)out_size; (void)ws_size;
  const float* e_q   = (const float*)d_in[0];
  const float* e_p   = (const float*)d_in[1];
  const float* f_s   = (const float*)d_in[2];
  const float* f_e   = (const float*)d_in[3];
  const float* Wih_f = (const float*)d_in[4];
  const float* Whh_f = (const float*)d_in[5];
  const float* bih_f = (const float*)d_in[6];
  const float* bhh_f = (const float*)d_in[7];
  const float* Wih_b = (const float*)d_in[8];
  const float* Whh_b = (const float*)d_in[9];
  const float* bih_b = (const float*)d_in[10];
  const float* bhh_b = (const float*)d_in[11];

  char* ws = (char*)d_ws;
  unsigned*       bar  = (unsigned*)(ws + BAR_OFF);
  float*          xbuf = (float*)(ws + X_OFF);
  unsigned short* Abf  = (unsigned short*)(ws + A_OFF);
  unsigned short* Bbf  = (unsigned short*)(ws + B_OFF);
  float*          WT   = (float*)(ws + WT_OFF);
  float*          gh   = (float*)(ws + GH_OFF);
  float*          out  = (float*)d_out;

  k_init<<<32, 256, 0, stream>>>(xbuf, bar);
  k_build_A<<<(TOKS * 2 * KP) / 256, 256, 0, stream>>>(e_q, e_p, f_s, f_e, Abf);
  k_build_B<<<(2 * NP * KP) / 256, 256, 0, stream>>>(Whh_f, Whh_b, Bbf);
  k_build_WT<<<(2 * 3 * HDIM * JSTRIDE) / 256, 256, 0, stream>>>(Wih_f, Wih_b, WT);
  k_gemm_gh<<<dim3(98, 36), 256, 0, stream>>>(Abf, Bbf, bhh_f, bhh_b, gh);
  k_scan<<<NBLK_SCAN, 256, 0, stream>>>(WT, gh, bih_f, bih_b,
                                        e_q, e_p, f_s, f_e, out, xbuf, bar);
}
